// SelfAttention_32023276159347
// MI455X (gfx1250) — compile-verified, matrix operand order checked by reading, not run on hardware
//
#include <hip/hip_runtime.h>
#include <hip/hip_fp16.h>

// ---------------------------------------------------------------------------
// Fused causal self-attention for MI455X (gfx1250, wave32, WMMA).
//   B=4, S=2048, DIM_IN=512, DIM_K=2048, DIM_V=64
// Plan:
//   kernel 1 (x3): Q/K/V = f16(x @ W)  via v_wmma_f32_16x16x32_f16
//   kernel 2     : flash attention (online softmax, causal) via WMMA,
//                  staging f16 Q/K tiles with GLOBAL_LOAD_ASYNC_TO_LDS_B128
// ---------------------------------------------------------------------------

typedef __attribute__((ext_vector_type(16))) _Float16 v16h;
typedef __attribute__((ext_vector_type(8)))  float    v8f;

// 16-byte packet type matching the async-to-LDS builtin's parameter type:
//   __attribute__((__vector_size__(4 * sizeof(int)))) int
typedef int b128i __attribute__((vector_size(16)));
typedef __attribute__((address_space(1))) b128i* gptr_b128;  // global
typedef __attribute__((address_space(3))) b128i* lptr_b128;  // LDS

#define BATCH   4
#define SEQ     2048
#define DIN     512
#define DK      2048
#define DV      64
#define ROWS    (BATCH * SEQ)          // 8192
#define SCALE_QK 0.022097086912079608f // 1/sqrt(2048)

#if __has_builtin(__builtin_amdgcn_global_load_async_to_lds_b128)
#define HAVE_ASYNC_LDS_COPY 1
#endif

__device__ __forceinline__ v8f v8f_zero() {
  v8f z = {0.f, 0.f, 0.f, 0.f, 0.f, 0.f, 0.f, 0.f};
  return z;
}

// 128-bit global -> LDS copy. On gfx1250 use the async-to-LDS path
// (tracked by ASYNCcnt); otherwise a vectorized VGPR round trip.
__device__ __forceinline__ void copy_b128_g2l(const _Float16* g, _Float16* l) {
#if HAVE_ASYNC_LDS_COPY
  __builtin_amdgcn_global_load_async_to_lds_b128(
      (gptr_b128)(b128i*)const_cast<_Float16*>(g),
      (lptr_b128)(b128i*)l, 0, 0);
#else
  *(uint4*)l = *(const uint4*)g;
#endif
}

__device__ __forceinline__ void copy_g2l_wait() {
#if HAVE_ASYNC_LDS_COPY
#if __has_builtin(__builtin_amdgcn_s_wait_asynccnt)
  __builtin_amdgcn_s_wait_asynccnt(0);
#else
  asm volatile("s_wait_asynccnt 0" ::: "memory");
#endif
#endif
}

// Load a 16(row) x 32(k) f16 fragment from a row-major LDS tile.
// Layout per CDNA5 ISA (16-bit A-matrix 16x32): lane L -> row = L&15,
// k-half = (L>>4)*8; VGPR j<4 holds K = 2j..2j+1 (+khalf), VGPR j>=4 holds
// K = 16 + 2(j-4)..+1 (+khalf).  Same layout reused for B by storing the
// B operand transposed in LDS (rows = N, cols = K).
__device__ __forceinline__ v16h load_frag16x32(const _Float16* base, int ld, int lane) {
  const int row = lane & 15;
  const int kh  = (lane >> 4) << 3;
  const _Float16* p = base + row * ld;
  v16h f;
#pragma unroll
  for (int j = 0; j < 4; ++j) {
    f[2 * j]     = p[kh + 2 * j];
    f[2 * j + 1] = p[kh + 2 * j + 1];
  }
#pragma unroll
  for (int j = 0; j < 4; ++j) {
    f[8 + 2 * j]     = p[16 + kh + 2 * j];
    f[8 + 2 * j + 1] = p[16 + kh + 2 * j + 1];
  }
  return f;
}

// ---------------------------------------------------------------------------
// Kernel 1: Out[ROWS, N] (f16) = X[ROWS, DIN] (f32) @ W[DIN, N] (f32)
// Block: 256 threads = 8 waves. Macro tile 128(M) x 64(N). K-chunks of 32.
// (f32 -> f16 conversion must pass through VGPRs, so staging is manual.)
// ---------------------------------------------------------------------------
__global__ __launch_bounds__(256) void proj_f16_kernel(
    const float* __restrict__ X,
    const float* __restrict__ W,
    _Float16* __restrict__ Out,
    int N)
{
  __shared__ alignas(16) _Float16 Xs[128][32];   // [m][k]
  __shared__ alignas(16) _Float16 Wt[64][32];    // [n][k]  (W chunk transposed)

  const int tid  = threadIdx.x;
  const int lane = tid & 31;
  const int wv   = tid >> 5;             // wave id 0..7 -> M sub-tile
  const int m0   = blockIdx.x * 128;
  const int n0   = blockIdx.y * 64;

  v8f acc[4];
#pragma unroll
  for (int t = 0; t < 4; ++t) acc[t] = v8f_zero();

  for (int kc = 0; kc < DIN; kc += 32) {
    __syncthreads();
    // stage X chunk (f32 -> f16)
    for (int e = tid; e < 128 * 32; e += 256) {
      const int r = e >> 5, k = e & 31;
      Xs[r][k] = (_Float16)X[(size_t)(m0 + r) * DIN + kc + k];
    }
    // stage W chunk transposed (coalesced along n)
    for (int e = tid; e < 64 * 32; e += 256) {
      const int n = e & 63, k = e >> 6;
      Wt[n][k] = (_Float16)W[(size_t)(kc + k) * N + n0 + n];
    }
    // prefetch next chunk into cache while we compute
    if (kc + 32 < DIN) {
      __builtin_prefetch(&X[(size_t)(m0 + (tid & 127)) * DIN + kc + 32], 0, 0);
      __builtin_prefetch(&W[(size_t)(kc + 32 + (tid & 31)) * N + n0], 0, 0);
    }
    __syncthreads();

    const v16h a = load_frag16x32(&Xs[wv * 16][0], 32, lane);
#pragma unroll
    for (int t = 0; t < 4; ++t) {
      const v16h b = load_frag16x32(&Wt[t * 16][0], 32, lane);
      acc[t] = __builtin_amdgcn_wmma_f32_16x16x32_f16(
          false, a, false, b, (short)0, acc[t], false, false);
    }
  }

  // C layout: element (vgpr r, lane) -> M = r + 8*(lane>>4), N = lane&15
  const int hf = lane >> 4;
  const int nl = lane & 15;
#pragma unroll
  for (int t = 0; t < 4; ++t) {
#pragma unroll
    for (int r = 0; r < 8; ++r) {
      const int row = m0 + wv * 16 + r + 8 * hf;
      const int col = n0 + t * 16 + nl;
      Out[(size_t)row * N + col] = (_Float16)acc[t][r];
    }
  }
}

// ---------------------------------------------------------------------------
// Kernel 2: causal flash attention.
// Grid: (SEQ/128, BATCH). Block: 256 threads = 8 waves; wave w owns query
// rows q0 + 16w .. +15. Key blocks of 64, processed only up to the diagonal.
// Q/K d-chunks are staged via async global->LDS b128 copies (no conversion).
// ---------------------------------------------------------------------------
__global__ __launch_bounds__(256) void flash_attn_kernel(
    const _Float16* __restrict__ Qh,   // [ROWS, DK]
    const _Float16* __restrict__ Kh,   // [ROWS, DK]
    const _Float16* __restrict__ Vh,   // [ROWS, DV]
    float* __restrict__ Out)           // [ROWS, DV]
{
  __shared__ alignas(16) _Float16 Qs[128][64];     // query d-chunk        (16 KB)
  __shared__ alignas(16) _Float16 Ks[64][64];      // key d-chunk          ( 8 KB)
  __shared__ alignas(16) _Float16 Vt[64][64];      // V^T: [dv][key]       ( 8 KB)
  __shared__ alignas(16) _Float16 Ps[8][16][64];   // per-wave P tiles     (16 KB)

  const int tid  = threadIdx.x;
  const int lane = tid & 31;
  const int wv   = tid >> 5;
  const int b    = blockIdx.y;
  const int q0   = blockIdx.x * 128;
  const size_t rowQ = (size_t)b * SEQ + q0;

  const int hf = lane >> 4;
  const int nl = lane & 15;

  v8f o[4];
#pragma unroll
  for (int t = 0; t < 4; ++t) o[t] = v8f_zero();
  float mrow[8], lrow[8];
#pragma unroll
  for (int r = 0; r < 8; ++r) { mrow[r] = -INFINITY; lrow[r] = 0.f; }

  const int nkb = q0 / 64 + 2;   // causal: key-block start <= q0+127

  for (int kb = 0; kb < nkb; ++kb) {
    const size_t rowK = (size_t)b * SEQ + kb * 64;

    __syncthreads();  // previous iteration done reading Vt
    // stage V^T for this key block (transpose -> element-wise, once per kb)
    for (int e = tid; e < 64 * 64; e += 256) {
      const int key = e >> 6, dv = e & 63;
      Vt[dv][key] = Vh[(rowK + key) * DV + dv];
    }

    // ---- scores S = Q . K^T over DK in chunks of 64 ----
    v8f s[4];
#pragma unroll
    for (int t = 0; t < 4; ++t) s[t] = v8f_zero();

    for (int dc = 0; dc < DK; dc += 64) {
      __syncthreads();
      // Q chunk: 128 rows x 64 f16 = 1024 b128 packets (4/thread)
#pragma unroll
      for (int e = tid; e < (128 * 64) / 8; e += 256) {
        const int r = e >> 3, seg = (e & 7) * 8;
        copy_b128_g2l(&Qh[(rowQ + r) * DK + dc + seg], &Qs[r][seg]);
      }
      // K chunk: 64 rows x 64 f16 = 512 b128 packets (2/thread)
#pragma unroll
      for (int e = tid; e < (64 * 64) / 8; e += 256) {
        const int r = e >> 3, seg = (e & 7) * 8;
        copy_b128_g2l(&Kh[(rowK + r) * DK + dc + seg], &Ks[r][seg]);
      }
      copy_g2l_wait();     // drain this wave's ASYNCcnt
      __syncthreads();     // make all waves' copies visible

#pragma unroll
      for (int kk = 0; kk < 2; ++kk) {
        const v16h a = load_frag16x32(&Qs[wv * 16][kk * 32], 64, lane);
#pragma unroll
        for (int t = 0; t < 4; ++t) {
          const v16h bb = load_frag16x32(&Ks[t * 16][kk * 32], 64, lane);
          s[t] = __builtin_amdgcn_wmma_f32_16x16x32_f16(
              false, a, false, bb, (short)0, s[t], false, false);
        }
      }
    }

    // ---- online softmax (per wave; row M = r + 8*hf across 16 lanes) ----
#pragma unroll
    for (int r = 0; r < 8; ++r) {
      const int qg = q0 + wv * 16 + r + 8 * hf;
      float rmax = -INFINITY;
#pragma unroll
      for (int t = 0; t < 4; ++t) {
        const int kg = kb * 64 + t * 16 + nl;
        float v = s[t][r] * SCALE_QK;
        v = (kg <= qg) ? v : -INFINITY;
        s[t][r] = v;
        rmax = fmaxf(rmax, v);
      }
      rmax = fmaxf(rmax, __shfl_xor(rmax, 1));
      rmax = fmaxf(rmax, __shfl_xor(rmax, 2));
      rmax = fmaxf(rmax, __shfl_xor(rmax, 4));
      rmax = fmaxf(rmax, __shfl_xor(rmax, 8));

      const float mnew  = fmaxf(mrow[r], rmax);
      const float alpha = __expf(mrow[r] - mnew);
      float rsum = 0.f;
#pragma unroll
      for (int t = 0; t < 4; ++t) {
        const float p = __expf(s[t][r] - mnew);
        rsum += p;
        Ps[wv][r + 8 * hf][t * 16 + nl] = (_Float16)p;
      }
      rsum += __shfl_xor(rsum, 1);
      rsum += __shfl_xor(rsum, 2);
      rsum += __shfl_xor(rsum, 4);
      rsum += __shfl_xor(rsum, 8);

      mrow[r] = mnew;
      lrow[r] = lrow[r] * alpha + rsum;
#pragma unroll
      for (int t = 0; t < 4; ++t) o[t][r] *= alpha;
    }

    __syncthreads();  // Ps and Vt consistent for the P@V pass

    // ---- O += P @ V  (contract over 64 keys = 2 WMMA k-steps) ----
#pragma unroll
    for (int kk = 0; kk < 2; ++kk) {
      const v16h a = load_frag16x32(&Ps[wv][0][kk * 32], 64, lane);
#pragma unroll
      for (int t = 0; t < 4; ++t) {
        const v16h bb = load_frag16x32(&Vt[t * 16][kk * 32], 64, lane);
        o[t] = __builtin_amdgcn_wmma_f32_16x16x32_f16(
            false, a, false, bb, (short)0, o[t], false, false);
      }
    }
  }

  // ---- epilogue: normalize and store f32 ----
#pragma unroll
  for (int t = 0; t < 4; ++t) {
#pragma unroll
    for (int r = 0; r < 8; ++r) {
      const int qg = q0 + wv * 16 + r + 8 * hf;
      Out[((size_t)b * SEQ + qg) * DV + t * 16 + nl] = o[t][r] / lrow[r];
    }
  }
}

// ---------------------------------------------------------------------------
// Host-side launch
// ---------------------------------------------------------------------------
extern "C" void kernel_launch(void* const* d_in, const int* in_sizes, int n_in,
                              void* d_out, int out_size, void* d_ws, size_t ws_size,
                              hipStream_t stream) {
  const float* x  = (const float*)d_in[0];
  const float* Wq = (const float*)d_in[1];
  const float* Wk = (const float*)d_in[2];
  const float* Wv = (const float*)d_in[3];
  float* out = (float*)d_out;

  // workspace: Qh | Kh | Vh  (f16) -> 2*33.55MB + 1.05MB ~= 68.2 MB
  _Float16* Qh = (_Float16*)d_ws;
  _Float16* Kh = Qh + (size_t)ROWS * DK;
  _Float16* Vh = Kh + (size_t)ROWS * DK;

  proj_f16_kernel<<<dim3(ROWS / 128, DK / 64), 256, 0, stream>>>(x, Wq, Qh, DK);
  proj_f16_kernel<<<dim3(ROWS / 128, DK / 64), 256, 0, stream>>>(x, Wk, Kh, DK);
  proj_f16_kernel<<<dim3(ROWS / 128, DV / 64), 256, 0, stream>>>(x, Wv, Vh, DV);

  flash_attn_kernel<<<dim3(SEQ / 128, BATCH), 256, 0, stream>>>(Qh, Kh, Vh, out);
}
